// SpeechEmbedder_85323820302788
// MI455X (gfx1250) — compile-verified, matrix operand order checked by reading, not run on hardware
//
#include <hip/hip_runtime.h>
#include <hip/hip_bf16.h>

// ---------------------------------------------------------------------------
// Problem constants
// ---------------------------------------------------------------------------
#define BATCH 512
#define TSEQ  180
#define NMELS 40
#define HID   768
#define GATES 3072   // 4*HID
#define PROJ  256

typedef __attribute__((ext_vector_type(16))) __bf16 v16bf;
typedef __attribute__((ext_vector_type(8)))  __bf16 v8bf;
typedef __attribute__((ext_vector_type(8)))  float  v8f;

union ABFrag { v16bf v; v8bf h[2]; };

// LDS A-tile row stride in elements (32 k-elems per tile * 8 tiles + 8 pad)
#define ASR 264

// ---------------------------------------------------------------------------
// Fragment loaders (CDNA5 WMMA bf16 16x16x32 layouts, wave32)
// A (16x32, row-major source): lane L -> row M = L%16; half = L/16.
//   elems 0..7  = K = half*8 + 0..7 ; elems 8..15 = K = 16 + half*8 + 0..7
// ---------------------------------------------------------------------------
__device__ __forceinline__ v16bf load_a_frag_ptr(const __bf16* p) {
    ABFrag u;
    u.h[0] = *(const v8bf*)(p);
    u.h[1] = *(const v8bf*)(p + 16);
    return u.v;
}

__device__ __forceinline__ float sigmoidf_fast(float x) {
    return 1.0f / (1.0f + __expf(-x));
}

// ---------------------------------------------------------------------------
// Weight packing: row-major fp32 [G][K] (split wa: k < Kpad, wb: k >= Kpad)
// -> WMMA-B-fragment-ordered bf16 (lane-contiguous 32B per lane per tile):
//   packed index = ((nt*KT + kt)*512 + lane*16 + e)
//   lane L: col N = L%16, half = L/16; elem e = 2v+p:
//   k_local = (v<4 ? 0 : 16) + half*8 + (v&3)*2 + p
// ---------------------------------------------------------------------------
__global__ void pack_w_kernel(const float* __restrict__ wa,
                              const float* __restrict__ wb,
                              __bf16* __restrict__ Wp,
                              int NT, int KT, int Kpad, int Kreal) {
    size_t id = (size_t)blockIdx.x * 256 + threadIdx.x;
    size_t total = (size_t)NT * KT * 512;
    if (id >= total) return;
    int e    = (int)(id & 15);
    int lane = (int)((id >> 4) & 31);
    size_t t2 = id >> 9;
    int kt = (int)(t2 % KT);
    int nt = (int)(t2 / KT);
    int nl = lane & 15, half = lane >> 4;
    int v = e >> 1, p = e & 1;
    int kl = ((v < 4) ? 0 : 16) + half * 8 + (v & 3) * 2 + p;
    int k = kt * 32 + kl;
    int g = nt * 16 + nl;
    float val;
    if (k < Kpad) val = (k < Kreal) ? wa[(size_t)g * Kreal + k] : 0.0f;
    else          val = wb[(size_t)g * HID + (k - Kpad)];
    Wp[id] = (__bf16)val;
}

__global__ void addbias_kernel(const float* __restrict__ a,
                               const float* __restrict__ b,
                               float* __restrict__ o) {
    int i = blockIdx.x * 256 + threadIdx.x;
    if (i < GATES) o[i] = a[i] + b[i];
}

// x fp32 [B][T][40] -> bf16 [T][B][64] (zero padded K 40..63)
__global__ void pack_x_kernel(const float* __restrict__ x,
                              __bf16* __restrict__ xb) {
    size_t id = (size_t)blockIdx.x * 256 + threadIdx.x;
    int k = (int)(id & 63);
    size_t t2 = id >> 6;
    int b = (int)(t2 % BATCH);
    int t = (int)(t2 / BATCH);
    if (t >= TSEQ) return;
    float v = (k < NMELS) ? x[((size_t)b * TSEQ + t) * NMELS + k] : 0.0f;
    xb[id] = (__bf16)v;
}

__global__ void zero_kernel(unsigned int* __restrict__ p, size_t n) {
    size_t i = (size_t)blockIdx.x * 256 + threadIdx.x;
    if (i < n) p[i] = 0u;
}

// ---------------------------------------------------------------------------
// Cooperative global -> LDS staging of an A chunk:
// 32 rows x (CHT*32) k-elems, LDS row stride ASR, 16B words spread over 256 thr
// ---------------------------------------------------------------------------
template <int CHT, int LDA>
__device__ __forceinline__ void stage_a(__bf16 (&As)[32 * ASR],
                                        const __bf16* __restrict__ src,
                                        int Mbase, int kbase, int tid) {
    constexpr int WORDS = 32 * CHT * 4;      // 16B words in chunk
    constexpr int PER_T = WORDS / 256;
#pragma unroll
    for (int j = 0; j < PER_T; ++j) {
        int wd  = tid + j * 256;
        int row = wd / (CHT * 4);
        int sg  = wd % (CHT * 4);
        v8bf d = *(const v8bf*)(src + (size_t)(Mbase + row) * LDA + kbase + sg * 8);
        *(v8bf*)(&As[row * ASR + sg * 8]) = d;
    }
}

// Compute CHT k-tiles out of the staged LDS chunk (A via DS, B via VMEM).
template <int CHT>
__device__ __forceinline__ void compute_chunk(const __bf16 (&As)[32 * ASR],
                                              const __bf16*& pB0,
                                              const __bf16*& pB1,
                                              v8f& acc00, v8f& acc01,
                                              v8f& acc10, v8f& acc11,
                                              int nl, int half) {
#pragma unroll
    for (int ktl = 0; ktl < CHT; ++ktl) {
        v16bf a0 = load_a_frag_ptr(&As[nl * ASR + ktl * 32 + half * 8]);
        v16bf a1 = load_a_frag_ptr(&As[(nl + 16) * ASR + ktl * 32 + half * 8]);
        v16bf b0 = *(const v16bf*)pB0;
        v16bf b1 = *(const v16bf*)pB1;
        __builtin_prefetch(pB0 + 512, 0, 1);
        acc00 = __builtin_amdgcn_wmma_f32_16x16x32_bf16(false, a0, false, b0, (short)0, acc00, false, false);
        acc01 = __builtin_amdgcn_wmma_f32_16x16x32_bf16(false, a0, false, b1, (short)0, acc01, false, false);
        acc10 = __builtin_amdgcn_wmma_f32_16x16x32_bf16(false, a1, false, b0, (short)0, acc10, false, false);
        acc11 = __builtin_amdgcn_wmma_f32_16x16x32_bf16(false, a1, false, b1, (short)0, acc11, false, false);
        pB0 += 512; pB1 += 512;
    }
}

// ---------------------------------------------------------------------------
// One LSTM time-step for one layer (fused input+recurrent GEMM).
// Grid: (BATCH/32, HID/64), block 256 (8 waves).
// Block tile: M=32, hid=64, all 4 gates. Wave w: gate=w&3, hsub=w>>2 (0..1).
// Each wave: 2(M) x 2(N) WMMA tiles; A staged through LDS (shared by all 8
// waves), B streamed from packed global weights.
// ---------------------------------------------------------------------------
template <int KT0, int KT1, int LDA0, int CH0, int CH1>
__global__ __launch_bounds__(256)
void lstm_step_kernel(const __bf16* __restrict__ Asrc0,   // x_t, ld = LDA0
                      const __bf16* __restrict__ Asrc1,   // h_{t-1}, ld = HID
                      const __bf16* __restrict__ Wp,
                      const float* __restrict__ bias,
                      float* __restrict__ c,
                      __bf16* __restrict__ hout) {
    constexpr int KTT = KT0 + KT1;
    __shared__ __align__(16) __bf16 As[32 * ASR];   // staged A chunk (16.5 KB)
    __shared__ float ldsg[4][64][33];               // [gate][hid_local][m_local]

    const int tid  = threadIdx.x;
    const int lane = tid & 31;
    const int w    = tid >> 5;
    const int gate = w & 3;
    const int hsub = w >> 2;                 // 0..1
    const int Mbase = blockIdx.x * 32;
    const int Hbase = blockIdx.y * 64;
    const int nl = lane & 15, half = lane >> 4;

    const int nt0 = gate * (HID / 16) + blockIdx.y * 4 + hsub * 2;

    v8f acc00, acc01, acc10, acc11;
#pragma unroll
    for (int i = 0; i < 8; ++i) {
        acc00[i] = 0.0f; acc01[i] = 0.0f; acc10[i] = 0.0f; acc11[i] = 0.0f;
    }

    const __bf16* pB0 = Wp + (size_t)nt0 * KTT * 512 + lane * 16;
    const __bf16* pB1 = pB0 + (size_t)KTT * 512;

    // ---- K part 1: x_t source ----
#pragma unroll
    for (int ch = 0; ch < KT0 / CH0; ++ch) {
        __syncthreads();
        stage_a<CH0, LDA0>(As, Asrc0, Mbase, ch * CH0 * 32, tid);
        __syncthreads();
        compute_chunk<CH0>(As, pB0, pB1, acc00, acc01, acc10, acc11, nl, half);
    }
    // ---- K part 2: h_{t-1} source ----
#pragma unroll
    for (int ch = 0; ch < KT1 / CH1; ++ch) {
        __syncthreads();
        stage_a<CH1, HID>(As, Asrc1, Mbase, ch * CH1 * 32, tid);
        __syncthreads();
        compute_chunk<CH1>(As, pB0, pB1, acc00, acc01, acc10, acc11, nl, half);
    }

    // C tile layout: lane holds col N=lane%16; VGPR r -> M = r + 8*(lane/16)
    const int h0 = hsub * 32 + nl;           // first n-tile's hid_local
#pragma unroll
    for (int r = 0; r < 8; ++r) {
        int mr = r + 8 * half;
        ldsg[gate][h0     ][mr     ] = acc00[r];
        ldsg[gate][h0 + 16][mr     ] = acc01[r];
        ldsg[gate][h0     ][mr + 16] = acc10[r];
        ldsg[gate][h0 + 16][mr + 16] = acc11[r];
    }
    __syncthreads();

    // Cell update: 32 (m) x 64 (hid) cells, 8 per thread, hid on lanes
#pragma unroll
    for (int j = 0; j < 8; ++j) {
        int idx = tid + j * 256;
        int hl = idx & 63;
        int ml = idx >> 6;
        int b_  = Mbase + ml;
        int hid = Hbase + hl;
        float xi = ldsg[0][hl][ml] + bias[hid];
        float xf = ldsg[1][hl][ml] + bias[HID + hid];
        float xg = ldsg[2][hl][ml] + bias[2 * HID + hid];
        float xo = ldsg[3][hl][ml] + bias[3 * HID + hid];
        float i_ = sigmoidf_fast(xi);
        float f_ = sigmoidf_fast(xf);
        float g_ = tanhf(xg);
        float o_ = sigmoidf_fast(xo);
        size_t ci = (size_t)b_ * HID + hid;
        float cn = f_ * c[ci] + i_ * g_;
        c[ci] = cn;
        float h = o_ * tanhf(cn);
        hout[ci] = (__bf16)h;
    }
}

// ---------------------------------------------------------------------------
// Projection GEMM: e[512][256] = h_last(bf16) @ proj_w^T + proj_b (fp32 out)
// Grid (16,4), block 256; wave w: msub=w>>2, nsub=w&3; K-tiles = 24.
// ---------------------------------------------------------------------------
__global__ __launch_bounds__(256)
void proj_kernel(const __bf16* __restrict__ hlast,
                 const __bf16* __restrict__ Wp,
                 const float* __restrict__ pb,
                 float* __restrict__ e) {
    const int tid  = threadIdx.x;
    const int lane = tid & 31;
    const int w    = tid >> 5;
    const int msub = w >> 2, nsub = w & 3;
    const int Mbase = blockIdx.x * 32 + msub * 16;
    const int ntile = blockIdx.y * 4 + nsub;
    const int nl = lane & 15, half = lane >> 4;

    v8f acc;
#pragma unroll
    for (int i = 0; i < 8; ++i) acc[i] = 0.0f;

    const __bf16* pA = hlast + (size_t)(Mbase + nl) * HID + half * 8;
    const __bf16* pB = Wp + (size_t)ntile * 24 * 512 + lane * 16;
#pragma unroll 4
    for (int kt = 0; kt < 24; ++kt) {
        v16bf a = load_a_frag_ptr(pA);
        v16bf b = *(const v16bf*)pB;
        acc = __builtin_amdgcn_wmma_f32_16x16x32_bf16(false, a, false, b,
                                                      (short)0, acc, false, false);
        pA += 32; pB += 512;
    }
    const int col = ntile * 16 + nl;
    const float bb = pb[col];
#pragma unroll
    for (int r = 0; r < 8; ++r) {
        int row = Mbase + r + 8 * half;
        e[(size_t)row * PROJ + col] = acc[r] + bb;
    }
}

// L2 normalize rows of e[512][256] -> out
__global__ __launch_bounds__(256)
void norm_kernel(const float* __restrict__ e, float* __restrict__ out) {
    __shared__ float s[256];
    int b = blockIdx.x, tid = threadIdx.x;
    float v = e[(size_t)b * PROJ + tid];
    s[tid] = v * v;
    __syncthreads();
    for (int st = 128; st > 0; st >>= 1) {
        if (tid < st) s[tid] += s[tid + st];
        __syncthreads();
    }
    out[(size_t)b * PROJ + tid] = v * rsqrtf(s[0]);
}

// ---------------------------------------------------------------------------
// Host launcher
// ---------------------------------------------------------------------------
extern "C" void kernel_launch(void* const* d_in, const int* in_sizes, int n_in,
                              void* d_out, int out_size, void* d_ws, size_t ws_size,
                              hipStream_t stream) {
    const float* x      = (const float*)d_in[0];
    const float* w_ih0  = (const float*)d_in[1];
    const float* w_hh0  = (const float*)d_in[2];
    const float* b_ih0  = (const float*)d_in[3];
    const float* b_hh0  = (const float*)d_in[4];
    const float* w_ih1  = (const float*)d_in[5];
    const float* w_hh1  = (const float*)d_in[6];
    const float* b_ih1  = (const float*)d_in[7];
    const float* b_hh1  = (const float*)d_in[8];
    const float* w_ih2  = (const float*)d_in[9];
    const float* w_hh2  = (const float*)d_in[10];
    const float* b_ih2  = (const float*)d_in[11];
    const float* b_hh2  = (const float*)d_in[12];
    const float* proj_w = (const float*)d_in[13];
    const float* proj_b = (const float*)d_in[14];
    float* out = (float*)d_out;

    // Workspace layout (all sizes multiples of 256B)
    constexpr size_t szHseq = (size_t)TSEQ * BATCH * HID * 2;   // 141,557,760
    constexpr size_t szXb   = (size_t)TSEQ * BATCH * 64 * 2;
    constexpr size_t szWp0  = (size_t)192 * 26 * 512 * 2;       // K=832
    constexpr size_t szWp12 = (size_t)192 * 48 * 512 * 2;       // K=1536
    constexpr size_t szWpP  = (size_t)16 * 24 * 512 * 2;        // proj
    constexpr size_t szBias = (size_t)GATES * 4;
    constexpr size_t szC    = (size_t)BATCH * HID * 4;
    constexpr size_t szHz   = (size_t)BATCH * HID * 2;

    char* p = (char*)d_ws;
    __bf16* hA   = (__bf16*)p;                 p += szHseq;
    __bf16* hB   = (__bf16*)p;                 p += szHseq;
    __bf16* xb   = (__bf16*)p;                 p += szXb;
    __bf16* Wp0  = (__bf16*)p;                 p += szWp0;
    __bf16* Wp1  = (__bf16*)p;                 p += szWp12;
    __bf16* Wp2  = (__bf16*)p;                 p += szWp12;
    __bf16* WpP  = (__bf16*)p;                 p += szWpP;
    float*  bias0 = (float*)p;                 p += szBias;
    float*  bias1 = (float*)p;                 p += szBias;
    float*  bias2 = (float*)p;                 p += szBias;
    float*  c0   = (float*)p;                  p += szC;
    float*  c1   = (float*)p;                  p += szC;
    float*  c2   = (float*)p;                  p += szC;
    __bf16* hz   = (__bf16*)p;                 p += szHz;
    float*  ebuf = (float*)p;                  p += (size_t)BATCH * PROJ * 4;

    // 1. Zero c0,c1,c2,hz (contiguous region)
    {
        size_t words = (3 * szC + szHz) / 4;
        zero_kernel<<<(unsigned)((words + 255) / 256), 256, 0, stream>>>((unsigned int*)c0, words);
    }
    // 2. Combined biases
    addbias_kernel<<<GATES / 256, 256, 0, stream>>>(b_ih0, b_hh0, bias0);
    addbias_kernel<<<GATES / 256, 256, 0, stream>>>(b_ih1, b_hh1, bias1);
    addbias_kernel<<<GATES / 256, 256, 0, stream>>>(b_ih2, b_hh2, bias2);
    // 3. Pack x
    {
        size_t total = (size_t)TSEQ * BATCH * 64;
        pack_x_kernel<<<(unsigned)(total / 256), 256, 0, stream>>>(x, xb);
    }
    // 4. Pack weights into WMMA-B order
    pack_w_kernel<<<(unsigned)((192ull * 26 * 512 + 255) / 256), 256, 0, stream>>>(
        w_ih0, w_hh0, Wp0, 192, 26, 64, NMELS);
    pack_w_kernel<<<(unsigned)((192ull * 48 * 512 + 255) / 256), 256, 0, stream>>>(
        w_ih1, w_hh1, Wp1, 192, 48, HID, HID);
    pack_w_kernel<<<(unsigned)((192ull * 48 * 512 + 255) / 256), 256, 0, stream>>>(
        w_ih2, w_hh2, Wp2, 192, 48, HID, HID);
    pack_w_kernel<<<(unsigned)((16ull * 24 * 512 + 255) / 256), 256, 0, stream>>>(
        proj_w, proj_w, WpP, 16, 24, HID, HID);

    // 5. LSTM layers: 3 x 180 sequential step kernels
    dim3 sgrid(BATCH / 32, HID / 64), sblk(256);
    for (int t = 0; t < TSEQ; ++t) {
        const __bf16* a0    = xb + (size_t)t * BATCH * 64;
        const __bf16* hprev = (t == 0) ? hz : hA + (size_t)(t - 1) * BATCH * HID;
        lstm_step_kernel<2, 24, 64, 2, 8><<<sgrid, sblk, 0, stream>>>(
            a0, hprev, Wp0, bias0, c0, hA + (size_t)t * BATCH * HID);
    }
    for (int t = 0; t < TSEQ; ++t) {
        const __bf16* a0    = hA + (size_t)t * BATCH * HID;
        const __bf16* hprev = (t == 0) ? hz : hB + (size_t)(t - 1) * BATCH * HID;
        lstm_step_kernel<24, 24, 768, 8, 8><<<sgrid, sblk, 0, stream>>>(
            a0, hprev, Wp1, bias1, c1, hB + (size_t)t * BATCH * HID);
    }
    for (int t = 0; t < TSEQ; ++t) {
        const __bf16* a0    = hB + (size_t)t * BATCH * HID;
        const __bf16* hprev = (t == 0) ? hz : hA + (size_t)(t - 1) * BATCH * HID;
        lstm_step_kernel<24, 24, 768, 8, 8><<<sgrid, sblk, 0, stream>>>(
            a0, hprev, Wp2, bias2, c2, hA + (size_t)t * BATCH * HID);
    }

    // 6. Projection + L2 normalize
    const __bf16* hlast = hA + (size_t)(TSEQ - 1) * BATCH * HID;
    proj_kernel<<<dim3(BATCH / 32, PROJ / 64), 256, 0, stream>>>(hlast, WpP, proj_b, ebuf);
    norm_kernel<<<BATCH, PROJ, 0, stream>>>(ebuf, out);

    (void)in_sizes; (void)n_in; (void)out_size; (void)ws_size;
}
